// ImageCaptionModel_61211873903371
// MI455X (gfx1250) — compile-verified, matrix operand order checked by reading, not run on hardware
//
#include <hip/hip_runtime.h>

// ---------------------------------------------------------------------------
// CDNA5 (gfx1250) bf16 WMMA implementation of a 2-layer GRU caption decoder.
// - All matmuls via v_wmma_f32_16x16x32_bf16 (fp32 accumulate).
// - Weights packed to bf16 transposed [N][K] so B tiles load like A tiles.
// - GRU gate math fused into GEMM epilogues (MODE is a template param so each
//   instantiation carries only its own epilogue -> no accumulator spills).
// - Logits batched over all 25 steps: [3200,512]x[512,10000] (57% of FLOPs).
// - Tile staging: global_load_async_to_lds_b128 (ASYNCcnt) + double-buffered
//   LDS, one workgroup barrier per K-chunk, WMMA overlapped with async copy.
// ---------------------------------------------------------------------------

typedef __attribute__((ext_vector_type(16))) __bf16 v16bf;
typedef __attribute__((ext_vector_type(8)))  float  v8f;
typedef unsigned short u16;

__device__ __forceinline__ u16 f2bf(float f) {
  unsigned u = __float_as_uint(f);
  u += 0x7FFFu + ((u >> 16) & 1u);   // round-to-nearest-even
  return (u16)(u >> 16);
}

union FragBF {
  v16bf v;
  uint4 q[2];
};

// Async global->LDS 16B copy, tracked by ASYNCcnt (gfx1250 VGLOBAL op 98).
__device__ __forceinline__ void async_copy_b128(unsigned ldsByteOff, const void* gsrc) {
  asm volatile("global_load_async_to_lds_b128 %0, %1, off"
               :: "v"(ldsByteOff), "v"((unsigned long long)(uintptr_t)gsrc)
               : "memory");
}
__device__ __forceinline__ void wait_async0() {
  asm volatile("s_wait_asynccnt 0x0" ::: "memory");
}

// ---------------------------------------------------------------------------
// WMMA GEMM: C[M,N] = A[M,K] * Bt[N,K] (both bf16 row-major), fp32 acc.
// 256 threads = 8 waves arranged 2x4. Wave tile = (BM/2) x (BN/4):
//   BM=BN=64  -> wave 32x16: MT=2, NT=1 (recurrent GEMMs, more WGs)
//   BM=BN=128 -> wave 64x32: MT=4, NT=2 (logits GEMM, 8 WMMA per chunk)
// Double-buffered LDS, async staging of chunk k+1 overlapped with chunk k.
// MODE: 1 leaky-relu bias, 2 sigmoid u|r (+bf16 r*h), 3 tanh+state update,
//       4 logits with [t,b]->[b,t] remap and N mask.
// ---------------------------------------------------------------------------
template <int BM, int BN, int MODE>
__global__ __launch_bounds__(256, 1) void gemm_wmma_bf16(
    const u16* __restrict__ A, int lda,
    const u16* __restrict__ B, int ldb,
    int K, int Nact, const float* __restrict__ bias,
    float* __restrict__ outF, int ldout,
    const float* __restrict__ Uu, const float* __restrict__ Hprev,
    float* __restrict__ Hout,
    u16* __restrict__ bf1, int ldbf1,
    u16* __restrict__ bf2, int ldbf2,
    u16* __restrict__ bf3, int ldbf3,
    int urHalf)
{
  constexpr int WM = BM / 2;              // wave M extent
  constexpr int WN = BN / 4;              // wave N extent
  constexpr int MT = WM / 16;             // wmma tiles per wave (M)
  constexpr int NT = WN / 16;             // wmma tiles per wave (N)
  constexpr int BUFU16 = (BM + BN) * 32;  // u16 per buffer (A tile + B tile)

  // Single arena so async-LDS byte offsets are well defined (base 0).
  __shared__ __align__(16) u16 smem[2 * BUFU16];

  const int tid   = threadIdx.x;
  const int lane  = tid & 31;
  const int wave  = tid >> 5;
  const int mBase = blockIdx.y * BM;
  const int nBase = blockIdx.x * BN;
  const int waveM = (wave >> 2) * WM;
  const int waveN = (wave & 3) * WN;

  const int l15  = lane & 15;
  const int lsel = (lane < 16) ? 0 : 8;   // K-half select (ISA 7.12.2 layout)

  v8f acc[MT][NT];
#pragma unroll
  for (int mt = 0; mt < MT; ++mt)
#pragma unroll
    for (int nt = 0; nt < NT; ++nt) acc[mt][nt] = (v8f){};

  // Stage one 32-wide K chunk of A and B tiles into buffer `sel` (async).
  auto stage = [&](int sel, int k0) {
    const unsigned base = (unsigned)(sel * BUFU16);
#pragma unroll
    for (int i = tid; i < BM * 4; i += 256) {       // A: BM rows x 32 cols
      int r = i >> 2, c = (i & 3) * 8;
      async_copy_b128((base + (unsigned)(r * 32 + c)) * 2u,
                      &A[(size_t)(mBase + r) * lda + (k0 + c)]);
    }
#pragma unroll
    for (int i = tid; i < BN * 4; i += 256) {       // Bt: BN rows x 32 cols
      int r = i >> 2, c = (i & 3) * 8;
      async_copy_b128((base + (unsigned)(BM * 32 + r * 32 + c)) * 2u,
                      &B[(size_t)(nBase + r) * ldb + (k0 + c)]);
    }
  };

  stage(0, 0);
  wait_async0();

  int curb = 0;
  for (int k0 = 0; k0 < K; k0 += 32) {
    __syncthreads();                             // buffer `curb` ready for all
    if (k0 + 32 < K) stage(curb ^ 1, k0 + 32);   // prefetch next chunk (async)

    const int aBase = curb * BUFU16;
    const int bBase = aBase + BM * 32;

    FragBF bf[NT];
#pragma unroll
    for (int nt = 0; nt < NT; ++nt) {
      int bi = bBase + (waveN + nt * 16 + l15) * 32 + lsel;
      bf[nt].q[0] = *(const uint4*)&smem[bi];
      bf[nt].q[1] = *(const uint4*)&smem[bi + 16];
    }
#pragma unroll
    for (int mt = 0; mt < MT; ++mt) {
      FragBF af;
      int ai = aBase + (waveM + mt * 16 + l15) * 32 + lsel;
      af.q[0] = *(const uint4*)&smem[ai];
      af.q[1] = *(const uint4*)&smem[ai + 16];
#pragma unroll
      for (int nt = 0; nt < NT; ++nt)
        acc[mt][nt] = __builtin_amdgcn_wmma_f32_16x16x32_bf16(
            false, af.v, false, bf[nt].v, (short)0, acc[mt][nt], false, false);
    }

    if (k0 + 32 < K) wait_async0();              // next chunk fully in LDS
    curb ^= 1;
  }

  // Epilogue. C layout: lane<16 -> col=l15, rows g..g+7; lane>=16 -> rows +8.
  const int rowHalf = (lane < 16) ? 0 : 8;
#pragma unroll
  for (int mt = 0; mt < MT; ++mt) {
#pragma unroll
    for (int nt = 0; nt < NT; ++nt) {
      const int col = nBase + waveN + nt * 16 + l15;
      if (col >= Nact) continue;
      const int rBase = mBase + waveM + mt * 16 + rowHalf;
#pragma unroll
      for (int g = 0; g < 8; ++g) {
        const int row = rBase + g;
        const float v = acc[mt][nt][g];
        if constexpr (MODE == 1) {               // leaky_relu(acc + bias)
          float y = v + bias[col];
          if (y < 0.f) y *= 0.01f;
          outF[(size_t)row * ldout + col] = y;
        } else if constexpr (MODE == 2) {        // u|r gates
          float y = 1.f / (1.f + __expf(-(v + bias[col])));
          if (col < urHalf) {
            outF[(size_t)row * urHalf + col] = y;        // u gate (fp32)
          } else {
            int c = col - urHalf;                        // r -> bf16(r*h_prev)
            float rh = y * Hprev[(size_t)row * urHalf + c];
            bf1[(size_t)row * ldbf1 + c] = f2bf(rh);
          }
        } else if constexpr (MODE == 3) {        // candidate + state update
          float hh = tanhf(v + bias[col]);
          float u  = Uu[(size_t)row * urHalf + col];
          float hp = Hprev[(size_t)row * urHalf + col];
          float hn = u * hp + (1.f - u) * hh;
          Hout[(size_t)row * urHalf + col] = hn;
          u16 hb = f2bf(hn);
          bf1[(size_t)row * ldbf1 + col] = hb;
          if (bf2) bf2[(size_t)row * ldbf2 + col] = hb;
          if (bf3) bf3[(size_t)row * ldbf3 + col] = hb;
        } else {                                 // MODE 4: logits, row=t*128+b
          float y = v + bias[col];
          int t  = row / 128;
          int bb = row % 128;
          outF[((size_t)bb * 25 + t) * 10000 + col] = y;
        }
      }
    }
  }
}

// ---------------------------------------------------------------------------
// Pack fp32 weight [Ksrc][Nsrc] -> bf16 transposed [Nd][Kd] (Kd==Ksrc),
// optional K reorder (x-part first, r*h-part last): src row =
//   (k < xlen) ? k + rhlen : k - xlen.  Rows n >= Nsrc are zero-filled.
// ---------------------------------------------------------------------------
__global__ void pack_wt(const float* __restrict__ src, int Ksrc, int Nsrc,
                        u16* __restrict__ dst, int Kd, int Nd,
                        int xlen, int rhlen)
{
  (void)Ksrc;
  int idx = blockIdx.x * 256 + threadIdx.x;
  if (idx >= Nd * Kd) return;
  int n = idx / Kd;
  int k = idx - n * Kd;
  int ks = (xlen > 0) ? ((k < xlen) ? (k + rhlen) : (k - xlen)) : k;
  float v = (n < Nsrc) ? src[(size_t)ks * Nsrc + n] : 0.f;
  dst[idx] = f2bf(v);
}

__global__ void f32_to_bf16_k(const float* __restrict__ s, u16* __restrict__ d, int n) {
  int i = blockIdx.x * 256 + threadIdx.x;
  if (i < n) d[i] = f2bf(s[i]);
}

__global__ void concat_bias(const float* __restrict__ a, const float* __restrict__ b,
                            float* __restrict__ d, int h) {
  int i = blockIdx.x * 256 + threadIdx.x;
  if (i < h)          d[i] = a[i];
  else if (i < 2 * h) d[i] = b[i - h];
}

__global__ void zero_u16(u16* p, int n) {
  int i = blockIdx.x * 256 + threadIdx.x;
  if (i < n) p[i] = 0;
}
__global__ void zero_f32(float* p, int n) {
  int i = blockIdx.x * 256 + threadIdx.x;
  if (i < n) p[i] = 0.f;
}

// Build X0all / XC0all columns 0..1023 = [bf16(emb[token]) | bf16(p)] for all t.
__global__ void prep_inputs(const int* __restrict__ tokens,
                            const float* __restrict__ emb,
                            const float* __restrict__ pF,
                            u16* __restrict__ X0all, u16* __restrict__ XC0all)
{
  int tb = blockIdx.x;                       // 0..3199 : t*128+b
  int j  = blockIdx.y * 256 + threadIdx.x;   // 0..1023
  int t = tb / 128, b = tb % 128;
  float v;
  if (j < 512) v = emb[(size_t)tokens[b * 25 + t] * 512 + j];
  else         v = pF[b * 512 + (j - 512)];
  u16 h = f2bf(v);
  size_t off = ((size_t)t * 128 + b) * 1536 + j;
  X0all[off]  = h;
  XC0all[off] = h;
}

__global__ void copy_hidden(const float* __restrict__ H0, const float* __restrict__ H1,
                            float* __restrict__ dst) {
  int i = blockIdx.x * 256 + threadIdx.x;
  if (i < 128 * 512) {
    dst[i]             = H0[i];
    dst[128 * 512 + i] = H1[i];
  }
}

// ---------------------------------------------------------------------------
// Host side
// ---------------------------------------------------------------------------
static inline int cdiv(int a, int b) { return (a + b - 1) / b; }

extern "C" void kernel_launch(void* const* d_in, const int* in_sizes, int n_in,
                              void* d_out, int out_size, void* d_ws, size_t ws_size,
                              hipStream_t stream) {
  (void)in_sizes; (void)n_in; (void)out_size; (void)ws_size;

  const int*   tokens = (const int*)  d_in[0];
  const float* cnn    = (const float*)d_in[1];
  const float* emb    = (const float*)d_in[2];
  const float* Win    = (const float*)d_in[3];
  const float* bin_   = (const float*)d_in[4];
  const float* Wout   = (const float*)d_in[5];
  const float* bout   = (const float*)d_in[6];
  const float* Wu0 = (const float*)d_in[7];  const float* bu0 = (const float*)d_in[8];
  const float* Wr0 = (const float*)d_in[9];  const float* br0 = (const float*)d_in[10];
  const float* Wc0 = (const float*)d_in[11]; const float* bc0 = (const float*)d_in[12];
  const float* Wu1 = (const float*)d_in[13]; const float* bu1 = (const float*)d_in[14];
  const float* Wr1 = (const float*)d_in[15]; const float* br1 = (const float*)d_in[16];
  const float* Wc1 = (const float*)d_in[17]; const float* bc1 = (const float*)d_in[18];

  // workspace layout
  char* w = (char*)d_ws;
  size_t off = 0;
  auto alloc = [&](size_t bytes) -> void* {
    void* p = w + off;
    off = (off + bytes + 255) & ~(size_t)255;
    return p;
  };
  u16* X0all  = (u16*)alloc((size_t)25 * 128 * 1536 * 2);  // [t][b][ emb|p | h0 ]
  u16* XC0all = (u16*)alloc((size_t)25 * 128 * 1536 * 2);  // [t][b][ emb|p | r*h0 ]
  u16* XH1    = (u16*)alloc((size_t)128 * 1024 * 2);       // [b][ h0 | h1 ]
  u16* XC1    = (u16*)alloc((size_t)128 * 1024 * 2);       // [b][ h0 | r*h1 ]
  u16* H1all  = (u16*)alloc((size_t)25 * 128 * 512 * 2);   // logits GEMM A
  u16* CNNbf  = (u16*)alloc((size_t)128 * 2048 * 2);
  u16* WINt   = (u16*)alloc((size_t)512 * 2048 * 2);
  u16* WUR0t  = (u16*)alloc((size_t)1024 * 1536 * 2);
  u16* WC0t   = (u16*)alloc((size_t)512 * 1536 * 2);
  u16* WUR1t  = (u16*)alloc((size_t)1024 * 1024 * 2);
  u16* WC1t   = (u16*)alloc((size_t)512 * 1024 * 2);
  u16* WOUTt  = (u16*)alloc((size_t)10112 * 512 * 2);      // padded to 79*128 rows
  float* pF   = (float*)alloc((size_t)128 * 512 * 4);
  float* U0   = (float*)alloc((size_t)128 * 512 * 4);
  float* U1   = (float*)alloc((size_t)128 * 512 * 4);
  float* H0   = (float*)alloc((size_t)128 * 512 * 4);
  float* H1   = (float*)alloc((size_t)128 * 512 * 4);
  float* BUR0 = (float*)alloc(1024 * 4);
  float* BUR1 = (float*)alloc(1024 * 4);

  const dim3 blk(256);

  // --- weight / input packing (recomputed every call; deterministic) -------
  f32_to_bf16_k<<<cdiv(128 * 2048, 256), blk, 0, stream>>>(cnn, CNNbf, 128 * 2048);
  pack_wt<<<cdiv(512 * 2048, 256), blk, 0, stream>>>(Win, 2048, 512, WINt, 2048, 512, 0, 0);
  pack_wt<<<cdiv(512 * 1536, 256), blk, 0, stream>>>(Wu0, 1536, 512, WUR0t,              1536, 512, 0, 0);
  pack_wt<<<cdiv(512 * 1536, 256), blk, 0, stream>>>(Wr0, 1536, 512, WUR0t + 512 * 1536, 1536, 512, 0, 0);
  pack_wt<<<cdiv(512 * 1536, 256), blk, 0, stream>>>(Wc0, 1536, 512, WC0t, 1536, 512, 1024, 512); // K order: [x | r*h]
  pack_wt<<<cdiv(512 * 1024, 256), blk, 0, stream>>>(Wu1, 1024, 512, WUR1t,              1024, 512, 0, 0);
  pack_wt<<<cdiv(512 * 1024, 256), blk, 0, stream>>>(Wr1, 1024, 512, WUR1t + 512 * 1024, 1024, 512, 0, 0);
  pack_wt<<<cdiv(512 * 1024, 256), blk, 0, stream>>>(Wc1, 1024, 512, WC1t, 1024, 512, 512, 512);
  pack_wt<<<cdiv(10112 * 512, 256), blk, 0, stream>>>(Wout, 512, 10000, WOUTt, 512, 10112, 0, 0);
  concat_bias<<<cdiv(1024, 256), blk, 0, stream>>>(bu0, br0, BUR0, 512);
  concat_bias<<<cdiv(1024, 256), blk, 0, stream>>>(bu1, br1, BUR1, 512);

  // --- state init ----------------------------------------------------------
  zero_u16<<<cdiv(128 * 1536, 256), blk, 0, stream>>>(X0all, 128 * 1536);  // t=0 slice (h0=0)
  zero_u16<<<cdiv(128 * 1024, 256), blk, 0, stream>>>(XH1, 128 * 1024);    // h1=0
  zero_u16<<<cdiv(128 * 1024, 256), blk, 0, stream>>>(XC1, 128 * 1024);
  zero_f32<<<cdiv(128 * 512, 256), blk, 0, stream>>>(H0, 128 * 512);
  zero_f32<<<cdiv(128 * 512, 256), blk, 0, stream>>>(H1, 128 * 512);

  // --- input layer: p = leaky_relu(cnn @ Win + bin) ------------------------
  gemm_wmma_bf16<64, 64, 1><<<dim3(512 / 64, 2), blk, 0, stream>>>(
      CNNbf, 2048, WINt, 2048, 2048, 512, bin_,
      pF, 512, nullptr, nullptr, nullptr,
      nullptr, 0, nullptr, 0, nullptr, 0, 512);

  // --- x inputs for all timesteps: [emb(tok) | p] --------------------------
  prep_inputs<<<dim3(25 * 128, 4), blk, 0, stream>>>(tokens, emb, pF, X0all, XC0all);

  // --- recurrence ----------------------------------------------------------
  for (int t = 0; t < 25; ++t) {
    u16* X0t  = X0all  + (size_t)t * 128 * 1536;
    u16* XC0t = XC0all + (size_t)t * 128 * 1536;

    // GRU0 u|r: sigmoid([x|h0] @ [Wu0|Wr0]); writes U0 and bf16(r*h0)->XC0t[:,1024:]
    gemm_wmma_bf16<64, 64, 2><<<dim3(1024 / 64, 2), blk, 0, stream>>>(
        X0t, 1536, WUR0t, 1536, 1536, 1024, BUR0,
        U0, 512, nullptr, H0, nullptr,
        XC0t + 1024, 1536, nullptr, 0, nullptr, 0, 512);

    // GRU0 candidate + update: h0 = u*h0 + (1-u)*tanh([x|r*h0] @ Wc0t + bc0)
    gemm_wmma_bf16<64, 64, 3><<<dim3(512 / 64, 2), blk, 0, stream>>>(
        XC0t, 1536, WC0t, 1536, 1536, 512, bc0,
        nullptr, 0, U0, H0, H0,
        XH1, 1024,                 // bf16 h0 -> GRU1 input [h0|h1]
        XC1, 1024,                 // bf16 h0 -> GRU1 candidate input
        (t < 24) ? (X0all + (size_t)(t + 1) * 128 * 1536 + 1024) : nullptr, 1536,
        512);

    // GRU1 u|r
    gemm_wmma_bf16<64, 64, 2><<<dim3(1024 / 64, 2), blk, 0, stream>>>(
        XH1, 1024, WUR1t, 1024, 1024, 1024, BUR1,
        U1, 512, nullptr, H1, nullptr,
        XC1 + 512, 1024, nullptr, 0, nullptr, 0, 512);

    // GRU1 candidate + update: writes H1 fp32, bf16(h1) -> XH1[:,512:] and H1all[t]
    gemm_wmma_bf16<64, 64, 3><<<dim3(512 / 64, 2), blk, 0, stream>>>(
        XC1, 1024, WC1t, 1024, 1024, 512, bc1,
        nullptr, 0, U1, H1, H1,
        XH1 + 512, 1024,
        H1all + (size_t)t * 128 * 512, 512,
        nullptr, 0, 512);
  }

  // --- batched logits GEMM: [25*128, 512] x [512, 10000] -------------------
  gemm_wmma_bf16<128, 128, 4><<<dim3(10112 / 128, 3200 / 128), blk, 0, stream>>>(
      H1all, 512, WOUTt, 512, 512, 10000, bout,
      (float*)d_out, 0, nullptr, nullptr, nullptr,
      nullptr, 0, nullptr, 0, nullptr, 0, 512);

  // --- hidden = stack([h0, h1]) after d_out[128*25*10000] ------------------
  copy_hidden<<<cdiv(128 * 512, 256), blk, 0, stream>>>(
      H0, H1, (float*)d_out + (size_t)128 * 25 * 10000);
}
